// DotRole_61710090109597
// MI455X (gfx1250) — matrix operand order repaired and checked by reading, not compiled, not run on hardware
//
#include <hip/hip_runtime.h>
#include <hip/hip_bf16.h>
#include <math.h>

// ---------------------------------------------------------------------------
// DotRole fused kernel for gfx1250 (MI455X), wave32 + V_WMMA_F32_16X16X32_BF16
// ---------------------------------------------------------------------------

typedef __bf16 bf16_t;
typedef __attribute__((ext_vector_type(16))) __bf16 v16bf;
typedef __attribute__((ext_vector_type(8)))  __bf16 v8bf;
typedef __attribute__((ext_vector_type(8)))  float  v8f;
typedef __attribute__((ext_vector_type(4)))  float  v4f;

#define B_TOT   16384
#define RNN     256
#define LAT     64
#define ATT     64
#define NA      32
#define HID     256
#define SLOPE   0.01f
#define TILE_B  16
#define NWAVES  4

// ---- workspace byte offsets (prep-kernel outputs) -------------------------
#define WS_W1H   0          // 256x256 bf16  (w1_h rows)         131072 B
#define WS_W2    131072     // 32x256  bf16  (msg_w2 rows)        16384 B
#define WS_QFC   147456     // 64x256  bf16  (q_fc_w rows)        32768 B
#define WS_KEYW  180224     // 64x256  bf16  (key_w rows)         32768 B
#define WS_AL    212992     // 32x64   bf16  (action_latent)       4096 B
#define WS_QRY   217088     // 32x64   bf16  (query = AL@qw^T+qb)  4096 B
#define WS_XA    221184     // 32x256  f32   (AL@w1_a^T + msg_b1) 32768 B
// total ~248 KB

// ---------------------------------------------------------------------------
// WMMA helpers
// ---------------------------------------------------------------------------
__device__ __forceinline__ v8f wmma_bf16(v16bf a, v16bf b, v8f c) {
  // D = A(16x32) * B(32x16) + C(16x16 f32)
  return __builtin_amdgcn_wmma_f32_16x16x32_bf16(
      /*neg_a=*/false, a, /*neg_b=*/false, b,
      /*c_mod=*/(short)0, c, /*reuse_a=*/false, /*reuse_b=*/false);
}

// A operand (16x32 bf16) from a row-major 16x256 bf16 LDS tile.
// ISA layout: lanes 0-15 -> row=lane, K = {koff..koff+7, 16+koff..16+koff+7},
// koff = 0 for lanes<16, 8 for lanes>=16.
__device__ __forceinline__ v16bf load_a_bf_lds(const bf16_t* base, int lane, int kbase) {
  const int m    = lane & 15;
  const int koff = (lane >> 4) << 3;               // 0 or 8
  const bf16_t* p0 = base + m * RNN + kbase + koff;
  v8bf lo = *(const v8bf*)p0;                      // 16B aligned
  v8bf hi = *(const v8bf*)(p0 + 16);
  v16bf a;
#pragma unroll
  for (int j = 0; j < 8; ++j) { a[j] = lo[j]; a[8 + j] = hi[j]; }
  return a;
}

// A operand from a row-major f32 LDS tile with scale + bf16 convert.
__device__ __forceinline__ v16bf load_a_f32_lds(const float* base, int stride,
                                                int lane, int kbase, float scale) {
  const int m    = lane & 15;
  const int koff = (lane >> 4) << 3;
  const float* p0 = base + m * stride + kbase + koff;
  v16bf a;
#pragma unroll
  for (int j = 0; j < 8; ++j) {
    a[j]     = (bf16_t)(p0[j]      * scale);
    a[8 + j] = (bf16_t)(p0[16 + j] * scale);
  }
  return a;
}

// B operand (32x16 bf16, K x N) from row-major global weights:
// lane n holds K of column n (== elems of row n), same order as A:
// K = {koff..koff+7, 16+koff..16+koff+7}, koff = 0 for lanes<16, 8 for lanes>=16.
__device__ __forceinline__ v16bf load_b_gbl(const bf16_t* base, int lane,
                                            int kbase, int stride) {
  const int n    = lane & 15;
  const int koff = (lane >> 4) << 3;               // 0 or 8
  const bf16_t* p = base + n * stride + kbase + koff;
  v8bf lo = *(const v8bf*)p;                       // 16B aligned
  v8bf hi = *(const v8bf*)(p + 16);
  v16bf b;
#pragma unroll
  for (int j = 0; j < 8; ++j) { b[j] = lo[j]; b[8 + j] = hi[j]; }
  return b;
}

// ---------------------------------------------------------------------------
// Prep kernel: bf16 weight casts + tiny action-side precomputes (run once)
// ---------------------------------------------------------------------------
__global__ void dotrole_prep(const float* __restrict__ al,      // 32x64
                             const float* __restrict__ q_fc_w,  // 64x256
                             const float* __restrict__ msg_w1,  // 256x320
                             const float* __restrict__ msg_b1,  // 256
                             const float* __restrict__ msg_w2,  // 32x256
                             const float* __restrict__ key_w,   // 64x256
                             const float* __restrict__ query_w, // 64x64
                             const float* __restrict__ query_b, // 64
                             bf16_t* __restrict__ w1h_bf, bf16_t* __restrict__ w2_bf,
                             bf16_t* __restrict__ qfc_bf, bf16_t* __restrict__ keyw_bf,
                             bf16_t* __restrict__ al_bf,  bf16_t* __restrict__ query_bf,
                             float*  __restrict__ xa) {
  const int t = threadIdx.x; // 256 threads
  for (int i = t; i < HID * RNN; i += 256) {       // w1_h = msg_w1[:, :256]
    int n = i >> 8, k = i & 255;
    w1h_bf[i] = (bf16_t)msg_w1[n * (RNN + LAT) + k];
  }
  for (int i = t; i < NA * HID;  i += 256) w2_bf[i]   = (bf16_t)msg_w2[i];
  for (int i = t; i < LAT * RNN; i += 256) qfc_bf[i]  = (bf16_t)q_fc_w[i];
  for (int i = t; i < ATT * RNN; i += 256) keyw_bf[i] = (bf16_t)key_w[i];
  for (int i = t; i < NA * LAT;  i += 256) al_bf[i]   = (bf16_t)al[i];
  // xa[a][j] = msg_b1[j] + sum_l al[a][l] * msg_w1[j][256+l]
  for (int i = t; i < NA * HID; i += 256) {
    int a = i >> 8, j = i & 255;
    float s = msg_b1[j];
    for (int l = 0; l < LAT; ++l)
      s += al[a * LAT + l] * msg_w1[j * (RNN + LAT) + RNN + l];
    xa[i] = s;
  }
  // query[a][t2] = query_b[t2] + sum_l al[a][l] * query_w[t2][l]
  for (int i = t; i < NA * ATT; i += 256) {
    int a = i >> 6, t2 = i & 63;
    float s = query_b[t2];
    for (int l = 0; l < LAT; ++l) s += al[a * LAT + l] * query_w[t2 * LAT + l];
    query_bf[i] = (bf16_t)s;
  }
}

// ---------------------------------------------------------------------------
// Main fused kernel: 1 block == 16 rows of h, 4 waves
// ---------------------------------------------------------------------------
__global__ void __launch_bounds__(128)
dotrole_main(const float* __restrict__ h,
             const float* __restrict__ q_fc_b, const float* __restrict__ key_b,
             const float* __restrict__ msg_b2,
             const bf16_t* __restrict__ w1h_bf, const bf16_t* __restrict__ w2_bf,
             const bf16_t* __restrict__ qfc_bf, const bf16_t* __restrict__ keyw_bf,
             const bf16_t* __restrict__ al_bf,  const bf16_t* __restrict__ query_bf,
             const float*  __restrict__ xa,
             float* __restrict__ out) {
  __shared__ __align__(16) bf16_t sh_h[TILE_B * RNN];     //  8 KB, row-major
  __shared__ __align__(16) float  sh_xh[TILE_B * RNN];    // 16 KB
  __shared__ __align__(16) float  sh_xa[NA * HID];        // 32 KB
  __shared__ __align__(16) float  sh_rk[TILE_B * LAT];    //  4 KB
  __shared__ __align__(16) float  sh_key[TILE_B * ATT];   //  4 KB
  __shared__ __align__(16) float  sh_q[TILE_B * NA];      //  2 KB
  __shared__ __align__(16) float  sh_sm[TILE_B * NA];     //  2 KB
  __shared__ __align__(16) float  sh_msg[NWAVES][TILE_B * NA]; // 8 KB

  const int tid  = threadIdx.x;
  const int wave = tid >> 5;
  const int lane = tid & 31;
  const int tile = blockIdx.x;                     // b-row tile

  // ---- stage h tile (f32 -> bf16) and xa into LDS -------------------------
  const float4* hsrc = (const float4*)(h + (size_t)tile * TILE_B * RNN);
  for (int i = tid; i < TILE_B * RNN / 4; i += 128) {
    float4 v = hsrc[i];
    int b0 = i * 4;
    sh_h[b0 + 0] = (bf16_t)v.x; sh_h[b0 + 1] = (bf16_t)v.y;
    sh_h[b0 + 2] = (bf16_t)v.z; sh_h[b0 + 3] = (bf16_t)v.w;
  }
  {
    const float4* xs = (const float4*)xa;
    float4* xd = (float4*)sh_xa;
    for (int i = tid; i < NA * HID / 4; i += 128) xd[i] = xs[i];
  }
  __syncthreads();

  const int mrow = (lane >> 4) << 3;               // 0 or 8 (C/D row base)
  const int ccol = lane & 15;                      // C/D column

  // ---- fused xh / role_key / key over h-tile ------------------------------
  // Each wave: 64-col slice of xh (4 tiles) + one 16-col tile of rk and key.
  // One A-operand load from LDS feeds 6 WMMAs per k-chunk.
  {
    const int nbase = wave * 64;                   // xh column base
    const int nb    = wave * 16;                   // rk/key column base
    v8f acc[4] = {v8f{}, v8f{}, v8f{}, v8f{}};
    v8f rk = v8f{}, ky = v8f{};
    for (int kc = 0; kc < 8; ++kc) {
      v16bf a = load_a_bf_lds(sh_h, lane, kc * 32);
#pragma unroll
      for (int nt = 0; nt < 4; ++nt) {
        v16bf b = load_b_gbl(w1h_bf + (nbase + nt * 16) * RNN, lane, kc * 32, RNN);
        acc[nt] = wmma_bf16(a, b, acc[nt]);
      }
      rk = wmma_bf16(a, load_b_gbl(qfc_bf  + nb * RNN, lane, kc * 32, RNN), rk);
      ky = wmma_bf16(a, load_b_gbl(keyw_bf + nb * RNN, lane, kc * 32, RNN), ky);
    }
#pragma unroll
    for (int nt = 0; nt < 4; ++nt)
#pragma unroll
      for (int r = 0; r < 8; ++r)
        sh_xh[(mrow + r) * RNN + nbase + nt * 16 + ccol] = acc[nt][r];
#pragma unroll
    for (int r = 0; r < 8; ++r) {
      sh_rk [(mrow + r) * LAT + nb + ccol] = rk[r] + q_fc_b[nb + ccol];
      sh_key[(mrow + r) * ATT + nb + ccol] = ky[r] + key_b[nb + ccol];
    }
  }
  __syncthreads();

  // ---- q = role_key @ AL^T  (wave 0) ;  scores = (key/8) @ query^T (wave 1)
  if (wave == 0) {
    v8f qa[2] = {v8f{}, v8f{}};
#pragma unroll
    for (int kc = 0; kc < 2; ++kc) {
      v16bf a = load_a_f32_lds(sh_rk, LAT, lane, kc * 32, 1.0f);
#pragma unroll
      for (int nt = 0; nt < 2; ++nt)
        qa[nt] = wmma_bf16(a, load_b_gbl(al_bf + nt * 16 * LAT, lane, kc * 32, LAT), qa[nt]);
    }
#pragma unroll
    for (int nt = 0; nt < 2; ++nt)
#pragma unroll
      for (int r = 0; r < 8; ++r)
        sh_q[(mrow + r) * NA + nt * 16 + ccol] = qa[nt][r];
  } else if (wave == 1) {
    v8f sc[2] = {v8f{}, v8f{}};
#pragma unroll
    for (int kc = 0; kc < 2; ++kc) {
      v16bf a = load_a_f32_lds(sh_key, ATT, lane, kc * 32, 0.125f); // 1/sqrt(64)
#pragma unroll
      for (int nt = 0; nt < 2; ++nt)
        sc[nt] = wmma_bf16(a, load_b_gbl(query_bf + nt * 16 * ATT, lane, kc * 32, ATT), sc[nt]);
    }
#pragma unroll
    for (int nt = 0; nt < 2; ++nt)
#pragma unroll
      for (int r = 0; r < 8; ++r)
        sh_sm[(mrow + r) * NA + nt * 16 + ccol] = sc[nt][r];
  }
  __syncthreads();

  // ---- softmax over 32 actions per row ------------------------------------
  if (tid < TILE_B) {
    float* row = sh_sm + tid * NA;
    float mx = row[0];
    for (int j = 1; j < NA; ++j) mx = fmaxf(mx, row[j]);
    float s = 0.f;
    for (int j = 0; j < NA; ++j) { float e = __expf(row[j] - mx); row[j] = e; s += e; }
    float inv = 1.f / s;
    for (int j = 0; j < NA; ++j) row[j] *= inv;
  }
  __syncthreads();

  // ---- msgsum = sum_a  leaky(xh + xa[a]) @ w2^T  (8 a-values per wave) ----
  // Loop order kc-outer / a-inner: w2 B-operands and the xh chunk are
  // loop-invariant across the 8 a-iterations -> hoisted into registers.
  {
    const int m    = lane & 15;
    const int koff = (lane >> 4) << 3;
    const float* xhrow = sh_xh + m * RNN;
    v8f macc[2] = {v8f{}, v8f{}};
    for (int kc = 0; kc < 8; ++kc) {
      const int k0 = kc * 32 + koff;
      // hoisted, invariant over the a-loop:
      const v16bf b0 = load_b_gbl(w2_bf,            lane, kc * 32, HID);
      const v16bf b1 = load_b_gbl(w2_bf + 16 * HID, lane, kc * 32, HID);
      v4f xh0a = *(const v4f*)(xhrow + k0);
      v4f xh0b = *(const v4f*)(xhrow + k0 + 4);
      v4f xh1a = *(const v4f*)(xhrow + k0 + 16);
      v4f xh1b = *(const v4f*)(xhrow + k0 + 20);
#pragma unroll
      for (int i = 0; i < 8; ++i) {
        const float* xarow = sh_xa + (wave * 8 + i) * HID;
        v4f xa0a = *(const v4f*)(xarow + k0);
        v4f xa0b = *(const v4f*)(xarow + k0 + 4);
        v4f xa1a = *(const v4f*)(xarow + k0 + 16);
        v4f xa1b = *(const v4f*)(xarow + k0 + 20);
        v16bf a;
#pragma unroll
        for (int j = 0; j < 4; ++j) {
          float v0 = xh0a[j] + xa0a[j];
          float v1 = xh0b[j] + xa0b[j];
          float v2 = xh1a[j] + xa1a[j];
          float v3 = xh1b[j] + xa1b[j];
          a[j]      = (bf16_t)(v0 >= 0.f ? v0 : SLOPE * v0);
          a[4 + j]  = (bf16_t)(v1 >= 0.f ? v1 : SLOPE * v1);
          a[8 + j]  = (bf16_t)(v2 >= 0.f ? v2 : SLOPE * v2);
          a[12 + j] = (bf16_t)(v3 >= 0.f ? v3 : SLOPE * v3);
        }
        macc[0] = wmma_bf16(a, b0, macc[0]);
        macc[1] = wmma_bf16(a, b1, macc[1]);
      }
    }
#pragma unroll
    for (int nt = 0; nt < 2; ++nt)
#pragma unroll
      for (int r = 0; r < 8; ++r)
        sh_msg[wave][(mrow + r) * NA + nt * 16 + ccol] = macc[nt][r];
  }
  __syncthreads();

  // ---- epilogue: out = q + sm * (sum_w msg_partial + 32*msg_b2) -----------
  for (int e = tid; e < TILE_B * NA; e += 128) {
    int o = e & 31;
    float ms = sh_msg[0][e] + sh_msg[1][e] + sh_msg[2][e] + sh_msg[3][e]
             + 32.0f * msg_b2[o];
    out[(size_t)tile * TILE_B * NA + e] = sh_q[e] + sh_sm[e] * ms;
  }
}

// ---------------------------------------------------------------------------
extern "C" void kernel_launch(void* const* d_in, const int* in_sizes, int n_in,
                              void* d_out, int out_size, void* d_ws, size_t ws_size,
                              hipStream_t stream) {
  const float* h_p   = (const float*)d_in[0];
  const float* al    = (const float*)d_in[1];
  const float* qfw   = (const float*)d_in[2];
  const float* qfb   = (const float*)d_in[3];
  const float* mw1   = (const float*)d_in[4];
  const float* mb1   = (const float*)d_in[5];
  const float* mw2   = (const float*)d_in[6];
  const float* mb2   = (const float*)d_in[7];
  const float* kw    = (const float*)d_in[8];
  const float* kb    = (const float*)d_in[9];
  const float* qw    = (const float*)d_in[10];
  const float* qb    = (const float*)d_in[11];
  float* out = (float*)d_out;

  char* ws = (char*)d_ws;
  bf16_t* w1h_bf   = (bf16_t*)(ws + WS_W1H);
  bf16_t* w2_bf    = (bf16_t*)(ws + WS_W2);
  bf16_t* qfc_bf   = (bf16_t*)(ws + WS_QFC);
  bf16_t* keyw_bf  = (bf16_t*)(ws + WS_KEYW);
  bf16_t* al_bf    = (bf16_t*)(ws + WS_AL);
  bf16_t* query_bf = (bf16_t*)(ws + WS_QRY);
  float*  xa       = (float*)(ws + WS_XA);

  dotrole_prep<<<1, 256, 0, stream>>>(al, qfw, mw1, mb1, mw2, kw, qw, qb,
                                      w1h_bf, w2_bf, qfc_bf, keyw_bf,
                                      al_bf, query_bf, xa);
  dotrole_main<<<B_TOT / TILE_B, NWAVES * 32, 0, stream>>>(
      h_p, qfb, kb, mb2, w1h_bf, w2_bf, qfc_bf, keyw_bf, al_bf, query_bf, xa, out);
}
